// Proc_72206990181060
// MI455X (gfx1250) — matrix operand order repaired and checked by reading, not compiled, not run on hardware
//
#include <hip/hip_runtime.h>

typedef _Float16 v8h  __attribute__((ext_vector_type(8)));
typedef _Float16 v16h __attribute__((ext_vector_type(16)));
typedef float    v8f  __attribute__((ext_vector_type(8)));

#define B_  16
#define K_  256
#define Z_  128
#define H_  128
#define M_  (B_ * K_)   // 4096 rows
#define LDA 256          // fused A row: [ z (128) | m (128) ] in f16
#define LDB 128          // Bt row length (N) for all three GEMMs
#define N_  128          // output columns for all three GEMMs

// ---------------------------------------------------------------------------
// Single range-partitioned conversion kernel (exactly 2304 * 256 threads):
//   [0, 524288)          : z  f32 [4096,128] -> Ah[:,0:128] f16
//   [524288, 540672)     : W1 f32 [128,128]  -> W1t[k][n] f16 (transpose)
//   [540672, 557056)     : W2 f32 [128,128]  -> W2t[k][n] f16 (transpose)
//   [557056, 589824)     : Wu f32 [128,256]  -> Wut[k][n] f16 (transpose)
// ---------------------------------------------------------------------------
__global__ __launch_bounds__(256)
void cvt_all(const float* __restrict__ z,  const float* __restrict__ W1,
             const float* __restrict__ W2, const float* __restrict__ Wu,
             _Float16* __restrict__ Ah,  _Float16* __restrict__ W1t,
             _Float16* __restrict__ W2t, _Float16* __restrict__ Wut) {
    int idx = blockIdx.x * 256 + threadIdx.x;
    if (idx < M_ * Z_) {                                  // z -> Ah left half
        int row = idx >> 7, col = idx & 127;
        Ah[row * LDA + col] = (_Float16)z[idx];
    } else if (idx < M_ * Z_ + Z_ * Z_) {                 // W1 -> W1t
        int t = idx - M_ * Z_;
        int n = t >> 7, k = t & 127;
        W1t[k * Z_ + n] = (_Float16)W1[t];
    } else if (idx < M_ * Z_ + 2 * Z_ * Z_) {             // W2 -> W2t
        int t = idx - (M_ * Z_ + Z_ * Z_);
        int n = t >> 7, k = t & 127;
        W2t[k * Z_ + n] = (_Float16)W2[t];
    } else {                                              // Wu [128,256] -> Wut [256,128]
        int t = idx - (M_ * Z_ + 2 * Z_ * Z_);
        int n = t >> 8, k = t & 255;
        Wut[k * H_ + n] = (_Float16)Wu[t];
    }
}

// ---------------------------------------------------------------------------
// One wave per 16x16 tile; all leading dims compile-time so every load/store
// is base + immediate offset.
//   A frag (16-bit 16x32): lane = row (lane&15); two 16B chunks at
//     K = (lane>>4)*8 and 16+(lane>>4)*8.
//   B frag: lane = N col (lane&15); K = (lane>>4)*8 + 0..7 and 16+(lane>>4)*8 + 0..7.
//   C/D:    VGPR r -> M = r + 8*(lane>>4), N = lane&15.
// ---------------------------------------------------------------------------
template <int KTOT, bool RELU>
__device__ __forceinline__
void gemm_tile(const _Float16* __restrict__ A, const _Float16* __restrict__ Bt,
               const float* __restrict__ bias, float* __restrict__ out,
               int tm, int tn, int lane) {
    const int khalf = lane >> 4;
    const int l15   = lane & 15;

    const _Float16* Ap = A + (size_t)(tm * 16 + l15) * LDA + khalf * 8;
    const _Float16* Bp = Bt + tn * 16 + l15;

    v8f c = {};
#pragma unroll
    for (int kb = 0; kb < KTOT; kb += 32) {
        v8h alo = *(const v8h*)(Ap + kb);
        v8h ahi = *(const v8h*)(Ap + kb + 16);
        v16h a  = __builtin_shufflevector(alo, ahi,
                  0, 1, 2, 3, 4, 5, 6, 7, 8, 9, 10, 11, 12, 13, 14, 15);

        v16h b;
#pragma unroll
        for (int s = 0; s < 16; ++s) {
            int kk = (s < 8) ? (khalf * 8 + s) : (16 + khalf * 8 + (s - 8));
            b[s] = Bp[(size_t)(kb + kk) * LDB];
        }

        c = __builtin_amdgcn_wmma_f32_16x16x32_f16(
                /*neg_a=*/false, a, /*neg_b=*/false, b,
                /*c_mod=*/(short)0, c, /*reuse_a=*/false, /*reuse_b=*/false);
    }

    const float bv = bias[tn * 16 + l15];
    float* op = out + (size_t)(tm * 16 + khalf * 8) * N_ + tn * 16 + l15;
#pragma unroll
    for (int r = 0; r < 8; ++r) {
        float v = c[r] + bv;
        if (RELU) v = fmaxf(v, 0.0f);
        op[r * N_] = v;      // immediate offsets: r * 512 bytes
    }
}

// Pass 1: both Linear layers in one launch; blockIdx.z selects (W1,b1)/(W2,b2).
__global__ __launch_bounds__(32)
void wmma_gemm12(const _Float16* __restrict__ A,
                 const _Float16* __restrict__ B1t, const _Float16* __restrict__ B2t,
                 const float* __restrict__ b1, const float* __restrict__ b2,
                 float* __restrict__ m1z, float* __restrict__ m2z) {
    const _Float16* Bt = blockIdx.z ? B2t : B1t;
    const float* bias  = blockIdx.z ? b2  : b1;
    float* out         = blockIdx.z ? m2z : m1z;
    gemm_tile<128, false>(A, Bt, bias, out, blockIdx.x, blockIdx.y, threadIdx.x);
}

// Pass 3: out = relu(concat(z, m) @ Wu^T + bu), K = 256.
__global__ __launch_bounds__(32)
void wmma_gemm_u(const _Float16* __restrict__ A, const _Float16* __restrict__ Bt,
                 const float* __restrict__ bias, float* __restrict__ out) {
    gemm_tile<256, true>(A, Bt, bias, out, blockIdx.x, blockIdx.y, threadIdx.x);
}

// ---------------------------------------------------------------------------
// m[b,i,:] = relu(m1z[b,i,:] + max_{j : P[b,j,i]!=0} m2z[b,j,:])
// (max_j relu(a+x_j) == relu(a + max_j x_j); self-loops guarantee a non-empty
//  neighbor set). One block per (b,i); P column is blockIdx/j-uniform ->
//  scalar loads + uniform branch; m2z rows are coalesced 512B L2 reads.
// ---------------------------------------------------------------------------
__global__ __launch_bounds__(128)
void masked_max(const float* __restrict__ m1z, const float* __restrict__ m2z,
                const int* __restrict__ P, _Float16* __restrict__ Ah) {
    const int bid = blockIdx.x;          // b*K + i
    const int b   = bid >> 8;
    const int i   = bid & (K_ - 1);
    const int t   = threadIdx.x;         // z index

    const int*   Pcol = P + (size_t)b * K_ * K_ + i;   // P[b, j, i], stride K_
    const float* m2b  = m2z + (size_t)b * K_ * Z_;

    float acc = -__builtin_inff();
    for (int j = 0; j < K_; ++j) {
        if (Pcol[(size_t)j * K_] != 0)
            acc = fmaxf(acc, m2b[(size_t)j * Z_ + t]);
    }
    float v = m1z[(size_t)bid * Z_ + t] + acc;
    v = fmaxf(v, 0.0f);
    Ah[(size_t)bid * LDA + Z_ + t] = (_Float16)v;
}

// ---------------------------------------------------------------------------
extern "C" void kernel_launch(void* const* d_in, const int* in_sizes, int n_in,
                              void* d_out, int out_size, void* d_ws, size_t ws_size,
                              hipStream_t stream) {
    const float* z  = (const float*)d_in[0];
    const int*   P  = (const int*)d_in[1];
    const float* W1 = (const float*)d_in[2];
    const float* b1 = (const float*)d_in[3];
    const float* W2 = (const float*)d_in[4];
    const float* b2 = (const float*)d_in[5];
    const float* Wu = (const float*)d_in[6];
    const float* bu = (const float*)d_in[7];
    float* out = (float*)d_out;

    // workspace carve-up (~6.4 MB total)
    char* ws = (char*)d_ws;
    _Float16* Ah  = (_Float16*)ws;  ws += (size_t)M_ * LDA * sizeof(_Float16);  // 2 MB
    _Float16* W1t = (_Float16*)ws;  ws += (size_t)Z_ * Z_ * sizeof(_Float16);
    _Float16* W2t = (_Float16*)ws;  ws += (size_t)Z_ * Z_ * sizeof(_Float16);
    _Float16* Wut = (_Float16*)ws;  ws += (size_t)2 * Z_ * H_ * sizeof(_Float16);
    float*    m1z = (float*)ws;     ws += (size_t)M_ * Z_ * sizeof(float);      // 2 MB
    float*    m2z = (float*)ws;     ws += (size_t)M_ * Z_ * sizeof(float);      // 2 MB

    // 1) stage all f16 operands in one launch (exact coverage: 2304 * 256)
    const int cvt_total = M_ * Z_ + 2 * Z_ * Z_ + H_ * 2 * Z_;   // 589824
    cvt_all<<<cvt_total / 256, 256, 0, stream>>>(z, W1, W2, Wu, Ah, W1t, W2t, Wut);

    // 2) m1z = z @ W1^T + b1  and  m2z = z @ W2^T + b2 in one launch (K = 128)
    dim3 g12(M_ / 16, N_ / 16, 2);
    wmma_gemm12<<<g12, 32, 0, stream>>>(Ah, W1t, W2t, b1, b2, m1z, m2z);

    // 3) masked neighbor max -> m, written as f16 into Ah[:, 128:256]
    masked_max<<<M_, Z_, 0, stream>>>(m1z, m2z, P, Ah);

    // 4) out = relu(concat(z, m) @ Wu^T + bu)   (K = 256)
    dim3 gu(M_ / 16, N_ / 16);
    wmma_gemm_u<<<gu, 32, 0, stream>>>(Ah, Wut, bu, out);
}
